// Node_Classification_32220844654961
// MI455X (gfx1250) — compile-verified, matrix operand order checked by reading, not run on hardware
//
#include <hip/hip_runtime.h>

// Problem constants (match reference)
#define NN    200000
#define EE    6400000
#define HID   128
#define OUT_F 16
#define EPSV  1e-5f

typedef __attribute__((ext_vector_type(16))) __bf16 v16bf;
typedef __attribute__((ext_vector_type(8)))  float  v8f;

// ---------------------------------------------------------------------------
// WMMA tile helpers.  CDNA5 wave32 16-bit A-operand layout (ISA 7.12.2):
//   lanes 0-15  : M = lane,    halves 0..7 = K[kb+0..7],  halves 8..15 = K[kb+16..23]
//   lanes 16-31 : M = lane-16, halves 0..7 = K[kb+8..15], halves 8..15 = K[kb+24..31]
// i.e. per-lane base offset k0 = (lane>=16)?8:0, two contiguous 16B chunks.
// B operand uses the symmetric layout with lane -> column N.
// ---------------------------------------------------------------------------
union ABf16 { v16bf v; uint4 q[2]; __bf16 h[16]; };

__device__ __forceinline__ v16bf load_tile_bf16(const __bf16* __restrict__ rp, int kb) {
  ABf16 u;
  u.q[0] = *(const uint4*)(rp + kb);        // 8 bf16, 16B
  u.q[1] = *(const uint4*)(rp + kb + 16);   // 8 bf16, 16B
  return u.v;
}

__device__ __forceinline__ v16bf load_tile_f32(const float* __restrict__ rp, int kb) {
  ABf16 u;
  const float4 f0 = *(const float4*)(rp + kb);
  const float4 f1 = *(const float4*)(rp + kb + 4);
  const float4 f2 = *(const float4*)(rp + kb + 16);
  const float4 f3 = *(const float4*)(rp + kb + 20);
  u.h[0]=(__bf16)f0.x; u.h[1]=(__bf16)f0.y; u.h[2]=(__bf16)f0.z; u.h[3]=(__bf16)f0.w;
  u.h[4]=(__bf16)f1.x; u.h[5]=(__bf16)f1.y; u.h[6]=(__bf16)f1.z; u.h[7]=(__bf16)f1.w;
  u.h[8]=(__bf16)f2.x; u.h[9]=(__bf16)f2.y; u.h[10]=(__bf16)f2.z; u.h[11]=(__bf16)f2.w;
  u.h[12]=(__bf16)f3.x; u.h[13]=(__bf16)f3.y; u.h[14]=(__bf16)f3.z; u.h[15]=(__bf16)f3.w;
  return u.v;
}

// One wave computes D[16x16] = A[m0:m0+16, 0:128] * Bt[col0:col0+16, 0:128]^T
// (Bt stored transposed: Bt[n][k]).  K=128 -> 4 x v_wmma_f32_16x16x32_bf16.
template <bool A_IS_F32>
__device__ __forceinline__ v8f wmma_row_tile(const void* __restrict__ Abase,
                                             const __bf16* __restrict__ Bt,
                                             int m0, int col0, int lane) {
  const int mrow = lane & 15;
  const int k0   = (lane >> 4) << 3;        // 0 or 8
  v8f acc = {};
  const __bf16* brow = Bt + (size_t)(col0 + mrow) * HID + k0;
  if constexpr (A_IS_F32) {
    const float* arow = (const float*)Abase + (size_t)(m0 + mrow) * HID + k0;
#pragma unroll
    for (int kb = 0; kb < HID; kb += 32) {
      v16bf a = load_tile_f32(arow, kb);
      v16bf b = load_tile_bf16(brow, kb);
      acc = __builtin_amdgcn_wmma_f32_16x16x32_bf16(false, a, false, b, (short)0, acc, false, false);
    }
  } else {
    const __bf16* arow = (const __bf16*)Abase + (size_t)(m0 + mrow) * HID + k0;
#pragma unroll
    for (int kb = 0; kb < HID; kb += 32) {
      v16bf a = load_tile_bf16(arow, kb);
      v16bf b = load_tile_bf16(brow, kb);
      acc = __builtin_amdgcn_wmma_f32_16x16x32_bf16(false, a, false, b, (short)0, acc, false, false);
    }
  }
  return acc;
}

// ---------------------------------------------------------------------------
// CSR construction: degree histogram -> single-block scan -> bucket fill
// ---------------------------------------------------------------------------
__global__ void k_degree(const int* __restrict__ dst, int* __restrict__ deg) {
  int e = blockIdx.x * 256 + threadIdx.x;
  if (e < EE) atomicAdd(&deg[dst[e]], 1);
}

__global__ void __launch_bounds__(1024)
k_scan(const int* __restrict__ deg, int* __restrict__ rowptr,
       int* __restrict__ cursor, float* __restrict__ inv) {
  __shared__ int sh[1024];
  const int t = threadIdx.x;
  int running = 0;
  for (int base = 0; base < NN; base += 1024) {
    const int i = base + t;
    const int d = (i < NN) ? deg[i] : 0;
    sh[t] = d;
    __syncthreads();
    for (int off = 1; off < 1024; off <<= 1) {
      int add = (t >= off) ? sh[t - off] : 0;
      __syncthreads();
      sh[t] += add;
      __syncthreads();
    }
    if (i < NN) {
      const int excl = running + sh[t] - d;
      rowptr[i] = excl;
      cursor[i] = excl;
      inv[i] = 1.0f / (float)(d + 1);
    }
    running += sh[1023];
    __syncthreads();
  }
  if (t == 0) rowptr[NN] = running;
}

__global__ void k_build(const int* __restrict__ src, const int* __restrict__ dst,
                        int* __restrict__ cursor, int* __restrict__ col) {
  int e = blockIdx.x * 256 + threadIdx.x;
  if (e < EE) {
    int p = atomicAdd(&cursor[dst[e]], 1);
    col[p] = src[e];
  }
}

// ---------------------------------------------------------------------------
// Weight prep: convert to bf16 + transpose so B columns are contiguous in K
// ---------------------------------------------------------------------------
__global__ void k_wconv(const float* __restrict__ W_in, const float* __restrict__ Ws,
                        const float* __restrict__ W_out, __bf16* __restrict__ WtIn,
                        __bf16* __restrict__ WtL, __bf16* __restrict__ WtOut) {
  int i = blockIdx.x * 256 + threadIdx.x;
  if (i < HID * HID) {
    int k = i >> 7, n = i & 127;
    WtIn[n * HID + k] = (__bf16)W_in[k * HID + n];
  }
  if (i < 3 * HID * HID) {
    int l = i / (HID * HID), r = i % (HID * HID);
    int k = r >> 7, n = r & 127;
    WtL[(size_t)l * HID * HID + n * HID + k] = (__bf16)Ws[(size_t)l * HID * HID + k * HID + n];
  }
  if (i < HID * OUT_F) {
    int k = i >> 4, n = i & 15;
    WtOut[n * HID + k] = (__bf16)W_out[k * OUT_F + n];
  }
}

// ---------------------------------------------------------------------------
// Aggregation: one wave per dst node, gather CSR neighbors, fp32 accumulate,
// fuse (sum + self) * inv, emit bf16 row for the next GEMM's A operand.
// ---------------------------------------------------------------------------
__global__ void k_aggregate(const float* __restrict__ x, const int* __restrict__ rowptr,
                            const int* __restrict__ col, const float* __restrict__ inv,
                            __bf16* __restrict__ hpre) {
  const int lane = threadIdx.x & 31;
  const int node = blockIdx.x * 8 + (threadIdx.x >> 5);
  const int beg = rowptr[node];
  const int end = rowptr[node + 1];
  float4 a = ((const float4*)(x + (size_t)node * HID))[lane];   // self contribution
  float a0 = a.x, a1 = a.y, a2 = a.z, a3 = a.w;
  for (int e = beg; e < end; ++e) {
    const int s = col[e];
    if (e + 2 < end) {  // hide indirect-gather latency (global_prefetch_b8)
      const int sp = col[e + 2];
      __builtin_prefetch(x + (size_t)sp * HID + lane * 4, 0, 1);
    }
    const float4 v = ((const float4*)(x + (size_t)s * HID))[lane];
    a0 += v.x; a1 += v.y; a2 += v.z; a3 += v.w;
  }
  const float sc = inv[node];
  union { __bf16 h[4]; uint2 u; } o;
  o.h[0] = (__bf16)(a0 * sc); o.h[1] = (__bf16)(a1 * sc);
  o.h[2] = (__bf16)(a2 * sc); o.h[3] = (__bf16)(a3 * sc);
  *(uint2*)(hpre + (size_t)node * HID + lane * 4) = o.u;
}

// ---------------------------------------------------------------------------
// GEMM kernels (C/D layout: lane<16 -> row v, col lane; lane>=16 -> row 8+v)
// ---------------------------------------------------------------------------
__global__ void __launch_bounds__(256)
k_gemm_in(const float* __restrict__ nodes, const __bf16* __restrict__ WtIn,
          const float* __restrict__ b_in, float* __restrict__ x) {
  const int lane = threadIdx.x & 31;
  const int w    = threadIdx.x >> 5;
  const int m0   = blockIdx.x * 16;
  const int col0 = w * 16;
  v8f acc = wmma_row_tile<true>(nodes, WtIn, m0, col0, lane);
  const int c  = lane & 15;
  const int hi = lane >> 4;
  const float bias = b_in[col0 + c];
#pragma unroll
  for (int v = 0; v < 8; ++v)
    x[(size_t)(m0 + v + 8 * hi) * HID + col0 + c] = acc[v] + bias;
}

__global__ void __launch_bounds__(256)
k_gemm_layer(const __bf16* __restrict__ hpre, const __bf16* __restrict__ Wt,
             const float* __restrict__ bias, const float* __restrict__ gamma,
             const float* __restrict__ beta, float* __restrict__ xout) {
  __shared__ float sh[16][HID];
  const int lane = threadIdx.x & 31;
  const int w    = threadIdx.x >> 5;
  const int m0   = blockIdx.x * 16;
  const int col0 = w * 16;
  v8f acc = wmma_row_tile<false>(hpre, Wt, m0, col0, lane);
  const int c  = lane & 15;
  const int hi = lane >> 4;
  const float b = bias[col0 + c];
#pragma unroll
  for (int v = 0; v < 8; ++v) {
    float h = acc[v] + b;
    sh[v + 8 * hi][col0 + c] = fmaxf(h, 0.0f);   // ReLU before norm, per DGL
  }
  __syncthreads();
  // LayerNorm over 128 features: 16 threads per row, 8 elems/thread,
  // 16-lane shuffle tree (wave32: lanes 0-15 = even row, 16-31 = odd row).
  const int r  = threadIdx.x >> 4;
  const int c0 = (threadIdx.x & 15) * 8;
  float s = 0.f, ss = 0.f;
#pragma unroll
  for (int j = 0; j < 8; ++j) { float t = sh[r][c0 + j]; s += t; ss += t * t; }
#pragma unroll
  for (int off = 8; off >= 1; off >>= 1) {
    s  += __shfl_xor(s,  off, 16);
    ss += __shfl_xor(ss, off, 16);
  }
  const float mu  = s * (1.0f / HID);
  const float var = fmaxf(ss * (1.0f / HID) - mu * mu, 0.0f);
  const float rs  = rsqrtf(var + EPSV);
  float* orow = xout + (size_t)(m0 + r) * HID;
#pragma unroll
  for (int j = 0; j < 8; ++j) {
    const int cc = c0 + j;
    orow[cc] = (sh[r][cc] - mu) * rs * gamma[cc] + beta[cc];
  }
}

__global__ void __launch_bounds__(256)
k_gemm_out(const float* __restrict__ x, const __bf16* __restrict__ WtOut,
           const float* __restrict__ b_out, float* __restrict__ out) {
  const int lane = threadIdx.x & 31;
  const int w    = threadIdx.x >> 5;
  const int m0   = blockIdx.x * 128 + w * 16;
  if (m0 >= NN) return;                       // wave-uniform: EXEC stays all-ones
  v8f acc = wmma_row_tile<true>(x, WtOut, m0, 0, lane);
  const int c  = lane & 15;
  const int hi = lane >> 4;
  const float b = b_out[c];
#pragma unroll
  for (int v = 0; v < 8; ++v)
    out[(size_t)(m0 + v + 8 * hi) * OUT_F + c] = acc[v] + b;
}

// ---------------------------------------------------------------------------
extern "C" void kernel_launch(void* const* d_in, const int* in_sizes, int n_in,
                              void* d_out, int out_size, void* d_ws, size_t ws_size,
                              hipStream_t stream) {
  const float* nodes  = (const float*)d_in[0];
  const int*   src    = (const int*)d_in[1];
  const int*   dst    = (const int*)d_in[2];
  const float* W_in   = (const float*)d_in[3];
  const float* b_in   = (const float*)d_in[4];
  const float* Ws     = (const float*)d_in[5];
  const float* bs     = (const float*)d_in[6];
  const float* gammas = (const float*)d_in[7];
  const float* betas  = (const float*)d_in[8];
  const float* W_out  = (const float*)d_in[9];
  const float* b_out  = (const float*)d_in[10];
  float* out = (float*)d_out;

  // Scratch carve-out (~175 MB total)
  char* p = (char*)d_ws;
  auto carve = [&](size_t bytes) { char* r = p; p += (bytes + 255) & ~(size_t)255; return r; };
  int*    deg    = (int*)carve((size_t)NN * 4);
  int*    rowptr = (int*)carve((size_t)(NN + 1) * 4);
  int*    cursor = (int*)carve((size_t)NN * 4);
  float*  inv    = (float*)carve((size_t)NN * 4);
  int*    col    = (int*)carve((size_t)EE * 4);
  float*  x      = (float*)carve((size_t)NN * HID * 4);
  __bf16* hpre   = (__bf16*)carve((size_t)NN * HID * 2);
  __bf16* WtIn   = (__bf16*)carve((size_t)HID * HID * 2);
  __bf16* WtL    = (__bf16*)carve((size_t)3 * HID * HID * 2);
  __bf16* WtOut  = (__bf16*)carve((size_t)OUT_F * HID * 2);

  hipMemsetAsync(deg, 0, (size_t)NN * 4, stream);
  k_degree<<<EE / 256, 256, 0, stream>>>(dst, deg);
  k_scan<<<1, 1024, 0, stream>>>(deg, rowptr, cursor, inv);
  k_build<<<EE / 256, 256, 0, stream>>>(src, dst, cursor, col);
  k_wconv<<<(3 * HID * HID + 255) / 256, 256, 0, stream>>>(W_in, Ws, W_out, WtIn, WtL, WtOut);

  k_gemm_in<<<NN / 16, 256, 0, stream>>>(nodes, WtIn, b_in, x);
  for (int l = 0; l < 3; ++l) {
    k_aggregate<<<NN / 8, 256, 0, stream>>>(x, rowptr, col, inv, hpre);
    k_gemm_layer<<<NN / 16, 256, 0, stream>>>(hpre, WtL + (size_t)l * HID * HID,
                                              bs + l * HID, gammas + l * HID,
                                              betas + l * HID, x);
  }
  k_gemm_out<<<(NN + 127) / 128, 256, 0, stream>>>(x, WtOut, b_out, out);
}
